// build_rnn_77008763617684
// MI455X (gfx1250) — matrix operand order repaired and checked by reading, not compiled, not run on hardware
//
#include <hip/hip_runtime.h>

// ---------------------------------------------------------------------------
// SCNN forward for MI455X (gfx1250).
// Phase 1: vertical dual-direction scan (1-channel 9-tap conv along W), one
//          block per (b,h), carry row in LDS, writes pf[B][C][H][W] f32.
// Phase 2: horizontal dual-direction scan. Each step = GEMM
//          [co=128] x [ci*9=1152] x [b*w=4096] done with
//          v_wmma_f32_16x16x32_bf16 (bf16 A/B, f32 accumulate).
//          Weights pre-packed in WMMA A-fragment layout; activations kept in
//          a transposed, zero-padded [b][w+4][ci] bf16 carry buffer so B
//          fragments are single contiguous 32B loads per lane.
// ---------------------------------------------------------------------------

typedef __attribute__((ext_vector_type(16))) __bf16 v16bf;
typedef __attribute__((ext_vector_type(8)))  float  v8f;

#define B_  16
#define C_  128
#define H_  64
#define W_  256
#define KW_ 9
#define WP_ 264   // W + 8 (4 zero-pad columns each side)

__device__ __forceinline__ unsigned short f2bf_bits(float f) {
  union { float f; unsigned u; } cv; cv.f = f;
  unsigned u = cv.u;
  unsigned r = (u + 0x7FFFu + ((u >> 16) & 1u)) >> 16;   // round-to-nearest-even
  return (unsigned short)r;
}

// ---------------------------------------------------------------------------
// Phase 1: vertical td (top->down over C) then dt (down->top), in-place in pf.
// pf[b][c][h][w] ends up holding the reference's "pf" tensor.
// ---------------------------------------------------------------------------
__global__ void scnn_vertical(const float* __restrict__ x,
                              const float* __restrict__ w1,
                              const float* __restrict__ b1,
                              float* __restrict__ pf) {
  __shared__ float row[W_ + 8];
  const int t  = threadIdx.x;          // w index, 256 threads
  const int bh = blockIdx.x;
  const int b  = bh >> 6;
  const int h  = bh & (H_ - 1);

  float cw[KW_];
#pragma unroll
  for (int k = 0; k < KW_; ++k) cw[k] = w1[k];
  const float bias = b1[0];

  if (t < 4) { row[t] = 0.f; row[W_ + 4 + t] = 0.f; }

  const size_t strideC = (size_t)H_ * W_;
  const size_t off0    = (((size_t)b * C_) * H_ + h) * W_ + t;

  float v0 = x[off0];
  row[4 + t] = v0;
  pf[off0]   = v0;
  __syncthreads();
  const float old0 = v0;

  // top -> down (td): td[1] = relu(c1(td[0])) + old[0]  (use_old1 == False)
  for (int c = 1; c < C_; ++c) {
    float conv = bias;
#pragma unroll
    for (int k = 0; k < KW_; ++k) conv = fmaf(cw[k], row[t + k], conv);
    float addv = (c == 1) ? old0 : x[off0 + (size_t)c * strideC];
    float v = fmaxf(conv, 0.f) + addv;
    __syncthreads();
    row[4 + t] = v;
    __syncthreads();
    pf[off0 + (size_t)c * strideC] = v;
  }

  // down -> top (dt), in place: pf[:,127-k] = relu(c1(carry)) + td[127-k]
  for (int k2 = 1; k2 < C_; ++k2) {
    const int c = C_ - 1 - k2;
    float conv = bias;
#pragma unroll
    for (int k = 0; k < KW_; ++k) conv = fmaf(cw[k], row[t + k], conv);
    float oldv = pf[off0 + (size_t)c * strideC];
    float v = fmaxf(conv, 0.f) + oldv;
    __syncthreads();
    row[4 + t] = v;
    __syncthreads();
    pf[off0 + (size_t)c * strideC] = v;
  }
}

// ---------------------------------------------------------------------------
// Pack conv2_w [co][ci][1][tap] f32 -> bf16 WMMA A-fragments.
// Fragment (ct, t, q): 32 lanes x 16 bf16; lane L holds M = L%16,
// element j -> Klocal = j + (L>=16 ? 16 : 0); ci = q*32+Klocal.
// ---------------------------------------------------------------------------
__global__ void scnn_pack_w2(const float* __restrict__ w2,
                             unsigned short* __restrict__ wpack) {
  const int idx  = blockIdx.x * 256 + threadIdx.x;   // < 8*9*4*32*16 = 147456
  const int j    = idx & 15;
  const int L    = (idx >> 4) & 31;
  const int frag = idx >> 9;                         // 0..287
  const int q    = frag & 3;
  const int t    = (frag >> 2) % KW_;
  const int ct   = frag / (4 * KW_);
  const int co   = ct * 16 + (L & 15);
  const int klocal = j + ((L >= 16) ? 16 : 0);
  const int ci   = q * 32 + klocal;
  wpack[idx] = f2bf_bits(w2[((size_t)co * C_ + ci) * KW_ + t]);
}

__global__ void scnn_zero(unsigned int* __restrict__ p, int n) {
  int i = blockIdx.x * 256 + threadIdx.x;
  if (i < n) p[i] = 0u;
}

// ---------------------------------------------------------------------------
// Seed a scan: build bf16 carry (transposed [b][w+4][c]) from pf slice h;
// optionally also emit the slice to out[b][w][c][h] (start of R->L pass).
// ---------------------------------------------------------------------------
__global__ void scnn_seed(const float* __restrict__ pf,
                          float* __restrict__ out,
                          unsigned short* __restrict__ carry_out,
                          int h, int write_out) {
  const int idx = blockIdx.x * 256 + threadIdx.x;    // B*C*W = 2^19
  const int w = idx & (W_ - 1);
  const int c = (idx >> 8) & (C_ - 1);
  const int b = idx >> 15;
  float v = pf[(((size_t)b * C_ + c) * H_ + h) * W_ + w];
  if (write_out) out[(((size_t)b * W_ + w) * C_ + c) * H_ + h] = v;
  carry_out[((size_t)b * WP_ + (w + 4)) * C_ + c] = f2bf_bits(v);
}

// ---------------------------------------------------------------------------
// One scan step: res = relu(GEMM(carry_in) + bias2) + pf[.,.,h,.]
// dst_mode 0: write res back into pf slice h (L->R pass, in-place)
// dst_mode 1: write res to out[b][w][c][h]     (R->L pass)
// Always emits bf16 carry_out in [b][w+4][c] layout for the next step.
// Block = 256 threads (8 waves); wave ct owns co tile [ct*16, ct*16+16).
// Grid = B * (W/16) = 256 blocks; each block covers one (b, 16-wide w tile).
// ---------------------------------------------------------------------------
__global__ void scnn_step(const unsigned short* __restrict__ carry_in,
                          unsigned short* __restrict__ carry_out,
                          const unsigned short* __restrict__ wpack,
                          const float* __restrict__ bias2,
                          float* __restrict__ pf,
                          float* __restrict__ out,
                          int h, int dst_mode) {
  const int lane = threadIdx.x & 31;
  const int ct   = threadIdx.x >> 5;             // 0..7 -> co tile
  const int b    = blockIdx.x >> 4;
  const int w0   = (blockIdx.x & 15) << 4;

  const int ncol  = lane & 15;                   // N (w within tile)
  const int khalf = (lane < 16) ? 0 : 16;        // B-fragment K half

  const unsigned short* wp = wpack + (size_t)ct * (KW_ * 4 * 512) + lane * 16;

  v8f acc = {};
#pragma unroll
  for (int t = 0; t < KW_; ++t) {
    // input column w' = w + t - 4  ->  padded index wp = w + t
    const unsigned short* brow =
        carry_in + ((size_t)b * WP_ + (w0 + ncol + t)) * C_ + khalf;
#pragma unroll
    for (int q = 0; q < 4; ++q) {
      v16bf a  = *reinterpret_cast<const v16bf*>(wp + (t * 4 + q) * 512);
      v16bf bb = *reinterpret_cast<const v16bf*>(brow + q * 32);
      acc = __builtin_amdgcn_wmma_f32_16x16x32_bf16(
          false, a, false, bb, (short)0, acc, false, false);
    }
  }

  // Epilog: lane holds N = w0+ncol; VGPR r holds M = r (+8 for upper lanes)
  const int w     = w0 + ncol;
  const int mbase = (lane < 16) ? 0 : 8;
  const int co0   = ct * 16 + mbase;
  unsigned int pk[4];
#pragma unroll
  for (int r = 0; r < 8; ++r) {
    const int co = co0 + r;
    float v = fmaxf(acc[r] + bias2[co], 0.f);
    const size_t aoff = (((size_t)b * C_ + co) * H_ + h) * W_ + w;
    float res = v + pf[aoff];
    if (dst_mode == 0) pf[aoff] = res;
    else out[(((size_t)b * W_ + w) * C_ + co) * H_ + h] = res;
    unsigned short bits = f2bf_bits(res);
    if (r & 1) pk[r >> 1] |= ((unsigned)bits) << 16;
    else       pk[r >> 1]  = bits;
  }
  uint4 v4; v4.x = pk[0]; v4.y = pk[1]; v4.z = pk[2]; v4.w = pk[3];
  *reinterpret_cast<uint4*>(
      carry_out + ((size_t)b * WP_ + (w + 4)) * C_ + co0) = v4;
}

// ---------------------------------------------------------------------------
// Workspace layout (bytes):
//   pf     @ 0           : B*C*H*W*4 = 134217728
//   wpack  @ 134217728   : 147456*2  = 294912
//   carry0 @ 134512640   : 16*264*128*2 = 1081344
//   carry1 @ 135593984   : 1081344       (total ~130.4 MB)
// ---------------------------------------------------------------------------
extern "C" void kernel_launch(void* const* d_in, const int* in_sizes, int n_in,
                              void* d_out, int out_size, void* d_ws, size_t ws_size,
                              hipStream_t stream) {
  (void)in_sizes; (void)n_in; (void)out_size; (void)ws_size;
  const float* x  = (const float*)d_in[0];
  const float* w1 = (const float*)d_in[1];
  const float* b1 = (const float*)d_in[2];
  const float* w2 = (const float*)d_in[3];
  const float* b2 = (const float*)d_in[4];
  float* out = (float*)d_out;

  char* ws = (char*)d_ws;
  float*          pf    = (float*)(ws);
  unsigned short* wpack = (unsigned short*)(ws + 134217728);
  unsigned short* c0    = (unsigned short*)(ws + 134512640);
  unsigned short* c1    = (unsigned short*)(ws + 135593984);
  unsigned short* cb[2] = {c0, c1};

  // Phase 1: vertical scans -> pf
  scnn_vertical<<<B_ * H_, 256, 0, stream>>>(x, w1, b1, pf);

  // Pack weights; zero carry buffers (pads must be 0 and stay 0)
  scnn_pack_w2<<<576, 256, 0, stream>>>(w2, wpack);
  scnn_zero<<<2112, 256, 0, stream>>>((unsigned int*)c0, 540672); // both buffers

  // Phase 2a: left->right over h (in-place in pf)
  scnn_seed<<<2048, 256, 0, stream>>>(pf, out, c0, 0, 0);
  for (int h = 1; h < H_; ++h)
    scnn_step<<<256, 256, 0, stream>>>(cb[(h + 1) & 1], cb[h & 1],
                                       wpack, b2, pf, out, h, 0);

  // Phase 2b: right->left over h (results permuted into out[b][w][c][h])
  scnn_seed<<<2048, 256, 0, stream>>>(pf, out, c0, H_ - 1, 1);
  for (int k = 1; k < H_; ++k)
    scnn_step<<<256, 256, 0, stream>>>(cb[(k + 1) & 1], cb[k & 1],
                                       wpack, b2, pf, out, H_ - 1 - k, 1);
}
